// VectorQuantizer_12524124635351
// MI455X (gfx1250) — matrix operand order repaired and checked by reading, not compiled, not run on hardware
//
#include <hip/hip_runtime.h>
#include <hip/hip_bf16.h>

// ---------------------------------------------------------------------------
// VQ quantizer for MI455X (gfx1250, wave32, WMMA).
// Distance GEMM done as split-bf16 (hi/lo) 3-pass WMMA with f32 accumulation:
// near-fp32 accuracy at bf16 matrix-core rate.
// ---------------------------------------------------------------------------

#define D_DIM   256
#define K_CODES 4096
#define N_ROWS  32768          // B*T = 16*2048
#define ROWS_PB 32             // rows per block in argmin kernel
#define COMMIT_F 0.25f

typedef __attribute__((ext_vector_type(16))) __bf16          v16bf;
typedef __attribute__((ext_vector_type(8)))  float           v8f;
typedef __attribute__((ext_vector_type(8)))  unsigned short  v8us;
typedef __attribute__((ext_vector_type(16))) unsigned short  v16us;

union FragA { v16bf bf; v8us  h[2]; };
union FragB { v16bf bf; v16us us;   };

__device__ __forceinline__ unsigned short f32_to_bf16(float f) {
    unsigned int u = __float_as_uint(f);
    u += 0x7FFFu + ((u >> 16) & 1u);        // round-to-nearest-even
    return (unsigned short)(u >> 16);
}
__device__ __forceinline__ float bf16_to_f32(unsigned short h) {
    return __uint_as_float(((unsigned int)h) << 16);
}

// ---------------------------------------------------------------------------
// Kernel 1: codebook -> bf16 hi/lo split + exact fp32 norms; zero counts.
// grid = K_CODES blocks, 256 threads (one code row per block).
// ---------------------------------------------------------------------------
__global__ __launch_bounds__(256) void vq_prep(
    const float* __restrict__ cb,
    float* __restrict__ cnorm,
    unsigned int* __restrict__ counts,
    unsigned short* __restrict__ cbh,
    unsigned short* __restrict__ cbl) {
    const int k = blockIdx.x;
    const int d = threadIdx.x;
    const float x = cb[k * D_DIM + d];
    const unsigned short h = f32_to_bf16(x);
    cbh[k * D_DIM + d] = h;
    cbl[k * D_DIM + d] = f32_to_bf16(x - bf16_to_f32(h));

    __shared__ float red[256];
    red[d] = x * x;
    __syncthreads();
    for (int s = 128; s > 0; s >>= 1) {
        if (d < s) red[d] += red[d + s];
        __syncthreads();
    }
    if (d == 0) cnorm[k] = red[0];
    if (d == 1) counts[k] = 0u;
}

// ---------------------------------------------------------------------------
// Kernel 2: argmin over K via WMMA. grid = N_ROWS/ROWS_PB blocks, 256 thr.
// score(n) = ||c_n||^2 - 2 * <z_m, c_n>   (row term constant, irrelevant)
// ---------------------------------------------------------------------------
__global__ __launch_bounds__(256) void vq_argmin(
    const float* __restrict__ z,
    const unsigned short* __restrict__ cbh,
    const unsigned short* __restrict__ cbl,
    const float* __restrict__ cnorm,
    int* __restrict__ idx_i,
    float* __restrict__ idx_f,
    unsigned int* __restrict__ counts) {

    __shared__ unsigned short As_hi[ROWS_PB * D_DIM];   // 16 KB
    __shared__ unsigned short As_lo[ROWS_PB * D_DIM];   // 16 KB
    __shared__ float candS[ROWS_PB][8];
    __shared__ int   candI[ROWS_PB][8];

    const int tid  = threadIdx.x;
    const int row0 = blockIdx.x * ROWS_PB;

    // Stage z tile into LDS as bf16 hi/lo (converted once, reused 256x).
    for (int i = tid; i < ROWS_PB * D_DIM; i += 256) {
        const float x = z[row0 * D_DIM + i];
        const unsigned short h = f32_to_bf16(x);
        As_hi[i] = h;
        As_lo[i] = f32_to_bf16(x - bf16_to_f32(h));
    }
    __syncthreads();

    const int wave = tid >> 5;
    const int lane = tid & 31;
    const int l15  = lane & 15;
    const int half = lane >> 4;

    for (int mtile = 0; mtile < ROWS_PB / 16; ++mtile) {
        float best[8];
        int   bidx[8];
#pragma unroll
        for (int e = 0; e < 8; ++e) { best[e] = 3.4e38f; bidx[e] = 0x7fffffff; }

        const int mA = mtile * 16 + l15;   // A-matrix row sourced by this lane

        for (int nt = wave; nt < K_CODES / 16; nt += 8) {
            const int n = nt * 16 + l15;   // B-matrix column sourced by this lane
            const unsigned short* __restrict__ bh = cbh + n * D_DIM;
            const unsigned short* __restrict__ bl = cbl + n * D_DIM;
            __builtin_prefetch(bh + 8 * 16 * D_DIM, 0, 1);   // next nt for this wave

            v8f acc = {};
#pragma unroll
            for (int ks = 0; ks < D_DIM / 32; ++ks) {
                // A 16x32 bf16 layout: lane half holds 16 contiguous K {h*16..h*16+15}
                const int c0 = ks * 32 + half * 16;
                FragA ah, al;
                ah.h[0] = *(const v8us*)&As_hi[mA * D_DIM + c0];
                ah.h[1] = *(const v8us*)&As_hi[mA * D_DIM + c0 + 8];
                al.h[0] = *(const v8us*)&As_lo[mA * D_DIM + c0];
                al.h[1] = *(const v8us*)&As_lo[mA * D_DIM + c0 + 8];
                // B 32x16 bf16 layout: lane half holds 16 contiguous K
                const int cb0 = ks * 32 + half * 16;
                FragB bhi, blo;
                bhi.us = *(const v16us*)&bh[cb0];
                blo.us = *(const v16us*)&bl[cb0];
                // split-precision accumulate: hi*hi + hi*lo + lo*hi
                acc = __builtin_amdgcn_wmma_f32_16x16x32_bf16(
                          false, ah.bf, false, bhi.bf, (short)0, acc, false, false);
                acc = __builtin_amdgcn_wmma_f32_16x16x32_bf16(
                          false, ah.bf, false, blo.bf, (short)0, acc, false, false);
                acc = __builtin_amdgcn_wmma_f32_16x16x32_bf16(
                          false, al.bf, false, bhi.bf, (short)0, acc, false, false);
            }

            const float cn = cnorm[n];
#pragma unroll
            for (int e = 0; e < 8; ++e) {
                // D layout: lane covers N=l15, M = half*8 + e
                const float s = cn - 2.0f * acc[e];
                if (s < best[e]) { best[e] = s; bidx[e] = n; }  // n monotone -> first-idx ties
            }
        }

        // Reduce across the 16 lanes sharing each row (flip l15 bits only).
#pragma unroll
        for (int e = 0; e < 8; ++e) {
            float s = best[e];
            int   i = bidx[e];
#pragma unroll
            for (int m = 1; m < 16; m <<= 1) {
                const float os = __shfl_xor(s, m, 32);
                const int   oi = __shfl_xor(i, m, 32);
                if (os < s || (os == s && oi < i)) { s = os; i = oi; }
            }
            if (l15 == 0) {
                const int rl = mtile * 16 + half * 8 + e;
                candS[rl][wave] = s;
                candI[rl][wave] = i;
            }
        }
    }
    __syncthreads();

    // Cross-wave reduction: one thread per row.
    if (tid < ROWS_PB) {
        float s = candS[tid][0];
        int   i = candI[tid][0];
        for (int w = 1; w < 8; ++w) {
            const float os = candS[tid][w];
            const int   oi = candI[tid][w];
            if (os < s || (os == s && oi < i)) { s = os; i = oi; }
        }
        const int grow = row0 + tid;
        idx_i[grow] = i;
        idx_f[grow] = (float)i;
        atomicAdd(&counts[i], 1u);      // integer atomics: order-independent
    }
}

// ---------------------------------------------------------------------------
// Kernel 3: z_q_st = codebook[idx] (STE forward value == z_q), per-row
// deterministic partial sums of (z_q - z)^2. grid = N_ROWS blocks.
// ---------------------------------------------------------------------------
__global__ __launch_bounds__(256) void vq_gather(
    const float* __restrict__ z,
    const float* __restrict__ cb,
    const int* __restrict__ idx_i,
    float* __restrict__ zq_out,
    float* __restrict__ lossPartial) {
    const int row = blockIdx.x;
    const int d   = threadIdx.x;
    const int k   = idx_i[row];
    const float q = cb[k * D_DIM + d];
    const float x = z[row * D_DIM + d];
    zq_out[row * D_DIM + d] = q;
    const float diff = q - x;

    __shared__ float red[256];
    red[d] = diff * diff;
    __syncthreads();
    for (int s = 128; s > 0; s >>= 1) {
        if (d < s) red[d] += red[d + s];
        __syncthreads();
    }
    if (d == 0) lossPartial[row] = red[0];
}

// ---------------------------------------------------------------------------
// Kernel 4: scalars. vq_loss = (1+COMMIT)*mean((zq-z)^2); perplexity; used.
// ---------------------------------------------------------------------------
__global__ __launch_bounds__(256) void vq_finalize(
    const float* __restrict__ lossPartial,
    const unsigned int* __restrict__ counts,
    float* __restrict__ out_scalars) {
    __shared__ float redL[256], redE[256], redU[256];
    const int t = threadIdx.x;
    float ls = 0.0f, ent = 0.0f, used = 0.0f;
    for (int i = t; i < N_ROWS; i += 256) ls += lossPartial[i];
    for (int k = t; k < K_CODES; k += 256) {
        const unsigned int c = counts[k];
        if (c > 0u) {
            const float p = (float)c * (1.0f / (float)N_ROWS);
            ent -= p * logf(p);
            used += 1.0f;
        }
    }
    redL[t] = ls; redE[t] = ent; redU[t] = used;
    __syncthreads();
    for (int s = 128; s > 0; s >>= 1) {
        if (t < s) { redL[t] += redL[t + s]; redE[t] += redE[t + s]; redU[t] += redU[t + s]; }
        __syncthreads();
    }
    if (t == 0) {
        out_scalars[0] = (1.0f + COMMIT_F) * redL[0] * (1.0f / ((float)N_ROWS * (float)D_DIM));
        out_scalars[1] = expf(redE[0]);
        out_scalars[2] = redU[0];
    }
}

// ---------------------------------------------------------------------------
// Workspace layout (bytes):             requirement ~= 4.3 MB
//   0        : cnorm        f32[4096]
//   16384    : counts       u32[4096]
//   32768    : idx_i        i32[32768]
//   163840   : lossPartial  f32[32768]
//   294912   : cb_hi        bf16[4096*256]   (2 MB)
//   2392064  : cb_lo        bf16[4096*256]   (2 MB)
// ---------------------------------------------------------------------------
extern "C" void kernel_launch(void* const* d_in, const int* in_sizes, int n_in,
                              void* d_out, int out_size, void* d_ws, size_t ws_size,
                              hipStream_t stream) {
    const float* z  = (const float*)d_in[0];   // [16,2048,256] f32
    const float* cb = (const float*)d_in[1];   // [4096,256]    f32

    char* ws = (char*)d_ws;
    float*          cnorm  = (float*)(ws + 0);
    unsigned int*   counts = (unsigned int*)(ws + 16384);
    int*            idx_i  = (int*)(ws + 32768);
    float*          lossP  = (float*)(ws + 163840);
    unsigned short* cbh    = (unsigned short*)(ws + 294912);
    unsigned short* cbl    = (unsigned short*)(ws + 294912 + 2097152);

    float* zq_out  = (float*)d_out;                       // 8388608 elems
    float* idx_f   = zq_out + (size_t)N_ROWS * D_DIM;     // 32768 elems
    float* scalars = idx_f + N_ROWS;                      // 3 elems

    vq_prep   <<<K_CODES,          256, 0, stream>>>(cb, cnorm, counts, cbh, cbl);
    vq_argmin <<<N_ROWS / ROWS_PB, 256, 0, stream>>>(z, cbh, cbl, cnorm, idx_i, idx_f, counts);
    vq_gather <<<N_ROWS,           256, 0, stream>>>(z, cb, idx_i, zq_out, lossP);
    vq_finalize<<<1,               256, 0, stream>>>(lossP, counts, scalars);
}
